// LSTM_63239098466711
// MI455X (gfx1250) — matrix operand order repaired and checked by reading, not compiled, not run on hardware
//
#include <hip/hip_runtime.h>
#include <hip/hip_bf16.h>
#include <math.h>

typedef __bf16 bf16_t;
typedef __attribute__((ext_vector_type(16))) __bf16 v16bf;
typedef __attribute__((ext_vector_type(8)))  __bf16 v8bf;
typedef __attribute__((ext_vector_type(8)))  float   v8f;

#define B_SZ 128
#define S_SZ 2048
#define D_SZ 128
#define H_SZ 512
#define C_SZ 10
#define KF   (D_SZ + H_SZ)   // 640 fused K (x | h)
#define NG   (4 * H_SZ)      // 2048 gate columns
#define NBLK 32              // 8 groups x 4 workgroups
#define NTHR 256             // 8 waves per WG; 32 waves per group = 32 hcol tiles
#define WGS_PER_GROUP 4
#define HGRP_ELEMS 8192      // 16 x 512 bf16 per buffer
#define HGRP_STRIDE 16384    // h0 + h1 per group, in elements

// ---------------- workspace layout (bytes, all 256-aligned) ----------------
#define WS_XBF   ((size_t)0)                               // B*S*D bf16 = 64 MiB
#define WS_WF    ((size_t)67108864)                        // NG*KF bf16 = 2.5 MiB
#define WS_H     (WS_WF + 2621440)                         // 8 groups * 32 KiB = 256 KiB
#define WS_BAR   (WS_H + 262144)                           // 8 barrier counters, 64B apart

__device__ __forceinline__ float sigmoid_fast(float x) {
    return 1.0f / (1.0f + __expf(-x));
}
__device__ __forceinline__ float tanh_fast(float x) {
    float cx = fminf(fmaxf(x, -15.0f), 15.0f);
    float e = __expf(2.0f * cx);
    return (e - 1.0f) / (e + 1.0f);
}

// Convert x [B,S,D] f32 -> bf16 (same layout). Exact grid: 33554432/4/256 = 32768 blocks.
__global__ __launch_bounds__(256) void cvt_x_kernel(const float* __restrict__ x,
                                                    bf16_t* __restrict__ xbf) {
    size_t i = ((size_t)blockIdx.x * 256 + threadIdx.x) * 4;
    const float4 v = *(const float4*)(x + i);
    xbf[i + 0] = (bf16_t)v.x;
    xbf[i + 1] = (bf16_t)v.y;
    xbf[i + 2] = (bf16_t)v.z;
    xbf[i + 3] = (bf16_t)v.w;
}

// Build fused, transposed, bf16 weight: Wf[n_global][k], n_global in [0,2048), k in [0,640).
// k<128 -> W_{G}x[k][n], else W_{G}h[k-128][n], where G = n_global/512, n = n_global%512.
// Exact grid: 2048*640/256 = 5120 blocks.
__global__ __launch_bounds__(256) void build_wf_kernel(
    const float* __restrict__ Wgx, const float* __restrict__ Wix,
    const float* __restrict__ Wfx, const float* __restrict__ Wox,
    const float* __restrict__ Wgh, const float* __restrict__ Wih,
    const float* __restrict__ Wfh, const float* __restrict__ Woh,
    bf16_t* __restrict__ Wf) {
    int idx = blockIdx.x * 256 + threadIdx.x;       // n_global*KF + k
    int n_global = idx / KF;
    int k = idx - n_global * KF;
    int G = n_global >> 9;
    int n = n_global & (H_SZ - 1);
    const float* Wx = (G == 0) ? Wgx : (G == 1) ? Wix : (G == 2) ? Wfx : Wox;
    const float* Wh = (G == 0) ? Wgh : (G == 1) ? Wih : (G == 2) ? Wfh : Woh;
    float v = (k < D_SZ) ? Wx[(size_t)k * H_SZ + n]
                         : Wh[(size_t)(k - D_SZ) * H_SZ + n];
    Wf[idx] = (bf16_t)v;
}

// Persistent LSTM scan. Batch rows are independent chains: 8 groups of 16 rows,
// each group = 4 workgroups (32 waves = 32 h-column tiles), private h buffers and
// a private 4-participant barrier. Each wave owns a 16x16 (rows x hcols) tile,
// computes all 4 gates with bf16 WMMA, keeps cell state c in registers.
__global__ __launch_bounds__(NTHR) void lstm_scan_kernel(
    const bf16_t* __restrict__ xbf, const bf16_t* __restrict__ Wf,
    const float* __restrict__ bgp, const float* __restrict__ bip,
    const float* __restrict__ bfp, const float* __restrict__ bop,
    bf16_t* __restrict__ H, unsigned* __restrict__ bar) {

    const int group  = blockIdx.x >> 2;                         // 0..7 (16 batch rows each)
    const int wgIn   = blockIdx.x & (WGS_PER_GROUP - 1);        // 0..3
    const int wave   = (wgIn << 3) | (threadIdx.x >> 5);        // 0..31 within group
    const int lane   = threadIdx.x & 31;
    const int laneLo = lane & 15;
    const int hi8    = lane >> 4;          // 0 or 1 (half-wave)
    const int nTile  = wave;               // 0..31 h-column tile
    const int bRow   = group * 16 + laneLo;      // global batch row this lane feeds (A)
    const int nCol   = nTile * 16 + laneLo;      // h column this lane feeds (B / output)

    // biases are per-column, invariant over t: hoist
    const float bg = bgp[nCol];
    const float bi = bip[nCol];
    const float bf = bfp[nCol];
    const float bo = bop[nCol];

    // per-lane base pointers
    const bf16_t* xrow0 = xbf + (size_t)bRow * (S_SZ * D_SZ);
    // pre-transposed fused weights: row (G*512 + nCol), B fragment chunk offset hi8*16
    const bf16_t* w0 = Wf + (size_t)(0 * H_SZ + nCol) * KF + hi8 * 16;
    const bf16_t* w1 = Wf + (size_t)(1 * H_SZ + nCol) * KF + hi8 * 16;
    const bf16_t* w2 = Wf + (size_t)(2 * H_SZ + nCol) * KF + hi8 * 16;
    const bf16_t* w3 = Wf + (size_t)(3 * H_SZ + nCol) * KF + hi8 * 16;

    // group-private h double buffer and barrier
    bf16_t* h0 = H + (size_t)group * HGRP_STRIDE;
    bf16_t* h1 = h0 + HGRP_ELEMS;
    unsigned* gbar = bar + group * 16;   // 64B apart

    v8f cacc = {};   // cell state for this tile, lives in registers across the scan

    for (int t = 0; t < S_SZ; ++t) {
        const bf16_t* hr = (t & 1) ? h1 : h0;       // read buffer
        bf16_t*       hw = (t & 1) ? h0 : h1;       // write buffer
        const bf16_t* xrow = xrow0 + (size_t)t * D_SZ;
        const bf16_t* hrow = hr + (size_t)laneLo * H_SZ;   // row-in-group

        // prefetch next timestep's x row for this lane into near caches (WGP scope):
        // x is the only HBM-sourced stream on the serial critical path.
        if (t + 1 < S_SZ) {
            __builtin_prefetch(xrow + D_SZ, 0, 3);
            __builtin_prefetch(xrow + D_SZ + 64, 0, 3);
        }

        v8f ag = {}, ai = {}, af = {}, ao = {};

        // x part of fused K: 4 blocks of 32, fully unrolled; the compiler hoists the
        // 16 loop-invariant x-weight fragments into registers across the whole scan.
        #pragma unroll
        for (int kb = 0; kb < 4; ++kb) {
            const int kw = kb * 32;
            const bf16_t* src = xrow + kw;
            v8bf aLo = *(const v8bf*)(src + hi8 * 16);
            v8bf aHi = *(const v8bf*)(src + hi8 * 16 + 8);
            v16bf afrag = __builtin_shufflevector(aLo, aHi,
                0, 1, 2, 3, 4, 5, 6, 7, 8, 9, 10, 11, 12, 13, 14, 15);
            v16bf b0 = *(const v16bf*)(w0 + kw);
            ag = __builtin_amdgcn_wmma_f32_16x16x32_bf16(false, afrag, false, b0, (short)0, ag, false, false);
            v16bf b1 = *(const v16bf*)(w1 + kw);
            ai = __builtin_amdgcn_wmma_f32_16x16x32_bf16(false, afrag, false, b1, (short)0, ai, false, false);
            v16bf b2 = *(const v16bf*)(w2 + kw);
            af = __builtin_amdgcn_wmma_f32_16x16x32_bf16(false, afrag, false, b2, (short)0, af, false, false);
            v16bf b3 = *(const v16bf*)(w3 + kw);
            ao = __builtin_amdgcn_wmma_f32_16x16x32_bf16(false, afrag, false, b3, (short)0, ao, false, false);
        }
        // h part of fused K: 16 blocks of 32 (weights stream from L2)
        #pragma unroll 4
        for (int kb = 0; kb < 16; ++kb) {
            const int kw = D_SZ + kb * 32;
            const bf16_t* src = hrow + kb * 32;
            v8bf aLo = *(const v8bf*)(src + hi8 * 16);
            v8bf aHi = *(const v8bf*)(src + hi8 * 16 + 8);
            v16bf afrag = __builtin_shufflevector(aLo, aHi,
                0, 1, 2, 3, 4, 5, 6, 7, 8, 9, 10, 11, 12, 13, 14, 15);
            v16bf b0 = *(const v16bf*)(w0 + kw);
            ag = __builtin_amdgcn_wmma_f32_16x16x32_bf16(false, afrag, false, b0, (short)0, ag, false, false);
            v16bf b1 = *(const v16bf*)(w1 + kw);
            ai = __builtin_amdgcn_wmma_f32_16x16x32_bf16(false, afrag, false, b1, (short)0, ai, false, false);
            v16bf b2 = *(const v16bf*)(w2 + kw);
            af = __builtin_amdgcn_wmma_f32_16x16x32_bf16(false, afrag, false, b2, (short)0, af, false, false);
            v16bf b3 = *(const v16bf*)(w3 + kw);
            ao = __builtin_amdgcn_wmma_f32_16x16x32_bf16(false, afrag, false, b3, (short)0, ao, false, false);
        }

        // epilogue: activations + cell/h update; C/D layout: VGPR v, lane -> (M = v + hi8*8, N = laneLo)
        #pragma unroll
        for (int v = 0; v < 8; ++v) {
            int Mrow = v + hi8 * 8;                 // row within group (0..15)
            float g  = tanh_fast(ag[v] + bg);
            float ii = sigmoid_fast(ai[v] + bi);
            float ff = sigmoid_fast(af[v] + bf);
            float oo = sigmoid_fast(ao[v] + bo);
            float cn = g * ii + cacc[v] * ff;
            cacc[v] = cn;
            hw[(size_t)Mrow * H_SZ + nCol] = (bf16_t)(tanh_fast(cn) * oo);
        }

        // group-private barrier (4 participants, monotonic counter on its own cacheline)
        __threadfence();
        __syncthreads();
        if (threadIdx.x == 0) {
            atomicAdd(gbar, 1u);
            const unsigned target = (unsigned)(t + 1) * WGS_PER_GROUP;
            while (*(volatile const unsigned*)gbar < target) {
                __builtin_amdgcn_s_sleep(2);
            }
        }
        __syncthreads();
        __threadfence();
    }
}

// Final projection: out[b][c] = sum_k h[b][k] * Wph[k][c] + bp[c]. 1280 outputs.
// Final h (t=2047, odd step) lives in each group's h0 buffer.
__global__ __launch_bounds__(256) void proj_kernel(const bf16_t* __restrict__ H,
                                                   const float* __restrict__ Wph,
                                                   const float* __restrict__ bp,
                                                   float* __restrict__ out) {
    int tid = blockIdx.x * 256 + threadIdx.x;
    if (tid >= B_SZ * C_SZ) return;
    int b = tid / C_SZ;
    int c = tid - b * C_SZ;
    float acc = bp[c];
    const bf16_t* hrow = H + (size_t)(b >> 4) * HGRP_STRIDE + (size_t)(b & 15) * H_SZ;
    #pragma unroll 8
    for (int k = 0; k < H_SZ; ++k) {
        acc = fmaf((float)hrow[k], Wph[(size_t)k * C_SZ + c], acc);
    }
    out[tid] = acc;
}

extern "C" void kernel_launch(void* const* d_in, const int* in_sizes, int n_in,
                              void* d_out, int out_size, void* d_ws, size_t ws_size,
                              hipStream_t stream) {
    const float* x   = (const float*)d_in[0];
    const float* Wgx = (const float*)d_in[1];
    const float* Wgh = (const float*)d_in[2];
    const float* bg  = (const float*)d_in[3];
    const float* Wix = (const float*)d_in[4];
    const float* Wih = (const float*)d_in[5];
    const float* bi  = (const float*)d_in[6];
    const float* Wfx = (const float*)d_in[7];
    const float* Wfh = (const float*)d_in[8];
    const float* bf  = (const float*)d_in[9];
    const float* Wox = (const float*)d_in[10];
    const float* Woh = (const float*)d_in[11];
    const float* bo  = (const float*)d_in[12];
    const float* Wph = (const float*)d_in[13];
    const float* bp  = (const float*)d_in[14];
    float* out = (float*)d_out;

    char* ws = (char*)d_ws;
    bf16_t*   xbf = (bf16_t*)(ws + WS_XBF);
    bf16_t*   Wf  = (bf16_t*)(ws + WS_WF);
    bf16_t*   H   = (bf16_t*)(ws + WS_H);
    unsigned* bar = (unsigned*)(ws + WS_BAR);

    // h0 of each group is read (as zeros) at t=0; barrier counters must start at 0 each call
    hipMemsetAsync(H, 0, 262144, stream);
    hipMemsetAsync(bar, 0, 1024, stream);

    cvt_x_kernel<<<32768, 256, 0, stream>>>(x, xbf);
    build_wf_kernel<<<5120, 256, 0, stream>>>(Wgx, Wix, Wfx, Wox, Wgh, Wih, Wfh, Woh, Wf);
    lstm_scan_kernel<<<NBLK, NTHR, 0, stream>>>(xbf, Wf, bg, bi, bf, bo, H, bar);
    proj_kernel<<<5, 256, 0, stream>>>(H, Wph, bp, out);
}
